// TransformerBlock_66589172957232
// MI455X (gfx1250) — matrix operand order repaired and checked by reading, not compiled, not run on hardware
//
#include <hip/hip_runtime.h>

// ---------------- problem constants ----------------
#define BB   4
#define TT   4096
#define DD   1024
#define HH   4096          // 4*D
#define MM   (BB*TT)       // 16384 rows
#define TC   128           // scan chunk (timesteps)
#define NC   (TT/TC)       // 32 chunks
#define EPSF 1e-5f

typedef __bf16 bf16;
typedef __attribute__((ext_vector_type(8)))  __bf16 bf16x8;
typedef __attribute__((ext_vector_type(16))) __bf16 v16bf;
typedef __attribute__((ext_vector_type(8)))  float  v8f;

// ---- gfx1250 async global->LDS DMA (ASYNCcnt path) ----
// INST_OFFSET is added to BOTH global and LDS addresses (ISA 08_async_tensor §4.4).
#define ASYNC_B128(ldsaddr, gaddr, off)                                          \
    asm volatile("global_load_async_to_lds_b128 %0, %1, off offset:" #off        \
                 :: "v"(ldsaddr), "v"(gaddr) : "memory")
#define WAIT_ASYNC(n) asm volatile("s_wait_asynccnt " #n ::: "memory")

__device__ __forceinline__ unsigned lds_addr(const void* p) {
    // generic shared pointer: low 32 bits are the LDS byte address
    return (unsigned)(unsigned long long)p;
}

__device__ __forceinline__ float geluf(float x) {
    return 0.5f * x * (1.0f + erff(x * 0.70710678118654752f));
}

// block-wide sum of a (float,float) pair; blockDim.x == 256 (8 wave32s)
__device__ __forceinline__ float2 block_reduce2(float a, float b, float2* sh) {
    #pragma unroll
    for (int o = 16; o > 0; o >>= 1) {
        a += __shfl_down(a, o, 32);
        b += __shfl_down(b, o, 32);
    }
    const int lane = threadIdx.x & 31;
    const int wid  = threadIdx.x >> 5;
    if (lane == 0) sh[wid] = make_float2(a, b);
    __syncthreads();
    if (wid == 0) {
        float2 t = (lane < 8) ? sh[lane] : make_float2(0.f, 0.f);
        a = t.x; b = t.y;
        #pragma unroll
        for (int o = 4; o > 0; o >>= 1) {
            a += __shfl_down(a, o, 32);
            b += __shfl_down(b, o, 32);
        }
        if (lane == 0) sh[8] = make_float2(a, b);
    }
    __syncthreads();
    float2 r = sh[8];
    __syncthreads();
    return r;
}

// ---------------- 1) transpose + fp32->bf16 cast ----------------
__global__ __launch_bounds__(256)
void transpose_cast_kernel(const float* __restrict__ W, bf16* __restrict__ Wt,
                           int R, int C) {
    __shared__ float tile[32][33];
    const int rb = blockIdx.y * 32, cb = blockIdx.x * 32;
    const int tx = threadIdx.x & 31, ty = threadIdx.x >> 5;   // 32x8
    #pragma unroll
    for (int i = 0; i < 32; i += 8)
        tile[ty + i][tx] = W[(size_t)(rb + ty + i) * C + (cb + tx)];
    __syncthreads();
    #pragma unroll
    for (int i = 0; i < 32; i += 8)
        Wt[(size_t)(cb + ty + i) * R + (rb + tx)] = (bf16)tile[tx][ty + i];
}

// ---------------- 2) LN1 + gelu + masked local cumsum ----------------
__global__ __launch_bounds__(256)
void ln1_scan_kernel(const float* __restrict__ x, const int* __restrict__ lengths,
                     const float* __restrict__ ln1w,
                     float* __restrict__ tmp1, float* __restrict__ chunksum) {
    __shared__ float2 sh[9];
    const int b = blockIdx.x, c = blockIdx.y, tid = threadIdx.x;
    const int len = lengths[b];
    float accv[4] = {0.f, 0.f, 0.f, 0.f};
    float w[4];
    #pragma unroll
    for (int j = 0; j < 4; ++j) w[j] = ln1w[tid + j * 256];

    const int t0 = c * TC;
    for (int t = t0; t < t0 + TC; ++t) {
        const size_t base = ((size_t)b * TT + t) * DD;
        float xv[4], s = 0.f, ss = 0.f;
        #pragma unroll
        for (int j = 0; j < 4; ++j) {
            xv[j] = x[base + tid + j * 256];
            s += xv[j]; ss += xv[j] * xv[j];
        }
        float2 r = block_reduce2(s, ss, sh);
        const float mu  = r.x * (1.f / DD);
        const float var = r.y * (1.f / DD) - mu * mu;
        const float rs  = rsqrtf(var + EPSF);
        const float vld = (t < len) ? 1.f : 0.f;
        #pragma unroll
        for (int j = 0; j < 4; ++j) {
            const float h = (xv[j] - mu) * rs * w[j];
            accv[j] += geluf(h) * vld;
            tmp1[base + tid + j * 256] = xv[j] + h + vld * accv[j];
        }
    }
    #pragma unroll
    for (int j = 0; j < 4; ++j)
        chunksum[((size_t)b * NC + c) * DD + tid + j * 256] = accv[j];
}

// ---------------- 3) exclusive prefix over chunks ----------------
__global__ __launch_bounds__(256)
void chunk_prefix_kernel(const float* __restrict__ chunksum,
                         float* __restrict__ chunkpref) {
    const int idx = blockIdx.x * 256 + threadIdx.x;   // B*D = 4096 lanes
    if (idx >= BB * DD) return;
    const int b = idx / DD, d = idx % DD;
    float run = 0.f;
    for (int c = 0; c < NC; ++c) {
        const size_t o = ((size_t)b * NC + c) * DD + d;
        chunkpref[o] = run;
        run += chunksum[o];
    }
}

// ---------------- 4) apply prefix (-> x1 in-place) + LN2 -> bf16 ----------------
__global__ __launch_bounds__(256)
void ln2_kernel(float* __restrict__ x1,
                const float* __restrict__ chunkpref,
                const int* __restrict__ lengths,
                const float* __restrict__ ln2w,
                bf16* __restrict__ h2) {
    __shared__ float2 sh[9];
    const int row = blockIdx.x;                 // 0 .. B*T-1
    const int b = row / TT, t = row % TT, c = t / TC;
    const float vld = (t < lengths[b]) ? 1.f : 0.f;
    const size_t base = (size_t)row * DD;
    const size_t pbase = ((size_t)b * NC + c) * DD;

    float v[4], s = 0.f, ss = 0.f;
    #pragma unroll
    for (int j = 0; j < 4; ++j) {
        const int d = threadIdx.x + j * 256;
        float val = x1[base + d] + vld * chunkpref[pbase + d];
        x1[base + d] = val;
        v[j] = val; s += val; ss += val * val;
    }
    float2 r = block_reduce2(s, ss, sh);
    const float mu  = r.x * (1.f / DD);
    const float var = r.y * (1.f / DD) - mu * mu;
    const float rs  = rsqrtf(var + EPSF);
    #pragma unroll
    for (int j = 0; j < 4; ++j) {
        const int d = threadIdx.x + j * 256;
        h2[base + d] = (bf16)((v[j] - mu) * rs * ln2w[d]);
    }
}

// ---------------- 5/6) bf16 WMMA GEMM, NT, async double-buffered LDS ----------------
// A: MxK row-major bf16 ; Bt: NxK row-major bf16
// EPI==0: C = gelu(A@B) stored bf16 ; EPI==1: C = A@B + resid stored f32
// 128x128 tile, K-stage = 64, 8 waves (2x4), wave tile 64x32 -> 16 WMMA per stage.
template <int EPI>
__global__ __launch_bounds__(256)
void gemm_bf16_kernel(const bf16* __restrict__ A, const bf16* __restrict__ Bt,
                      void* __restrict__ Cout, const float* __restrict__ resid,
                      int M, int N, int K) {
    __shared__ bf16 As[2][128][72];   // 64 K + 8 pad : row stride 144B (16B aligned)
    __shared__ bf16 Bs[2][128][72];   // total 72 KB

    const int m0   = blockIdx.x * 128;
    const int n0   = blockIdx.y * 128;
    const int tid  = threadIdx.x;
    const int lane = tid & 31;
    const int wave = tid >> 5;
    const int wm   = (wave & 1) * 64;    // wave row offset
    const int wn   = (wave >> 1) * 32;   // wave col offset
    const int l15  = lane & 15;
    const int hi   = lane >> 4;

    // loader mapping: 2 threads per row, each moves 64B (4 x b128) per tile
    const int ldr = tid >> 1;            // 0..127
    const int lc0 = (tid & 1) * 32;      // bf16 element offset: 0 or 32

    const bf16* gA = &A [(size_t)(m0 + ldr) * K + lc0];
    const bf16* gB = &Bt[(size_t)(n0 + ldr) * K + lc0];

    v8f acc[4][2];
    #pragma unroll
    for (int i = 0; i < 4; ++i)
        #pragma unroll
        for (int j = 0; j < 2; ++j)
            #pragma unroll
            for (int e = 0; e < 8; ++e) acc[i][j][e] = 0.f;

    const int S = K >> 6;   // stages of 64 K

    // issue stage 0 (8 async DMA loads per thread)
    {
        unsigned la = lds_addr(&As[0][ldr][lc0]);
        unsigned lb = lds_addr(&Bs[0][ldr][lc0]);
        unsigned long long ga = (unsigned long long)gA;
        unsigned long long gb = (unsigned long long)gB;
        ASYNC_B128(la, ga, 0);  ASYNC_B128(la, ga, 16);
        ASYNC_B128(la, ga, 32); ASYNC_B128(la, ga, 48);
        ASYNC_B128(lb, gb, 0);  ASYNC_B128(lb, gb, 16);
        ASYNC_B128(lb, gb, 32); ASYNC_B128(lb, gb, 48);
    }

    for (int s = 0; s < S; ++s) {
        const int cur = s & 1;
        if (s + 1 < S) {
            // prefetch next stage into the other buffer while this one computes
            unsigned la = lds_addr(&As[cur ^ 1][ldr][lc0]);
            unsigned lb = lds_addr(&Bs[cur ^ 1][ldr][lc0]);
            unsigned long long ga = (unsigned long long)(gA + (size_t)(s + 1) * 64);
            unsigned long long gb = (unsigned long long)(gB + (size_t)(s + 1) * 64);
            ASYNC_B128(la, ga, 0);  ASYNC_B128(la, ga, 16);
            ASYNC_B128(la, ga, 32); ASYNC_B128(la, ga, 48);
            ASYNC_B128(lb, gb, 0);  ASYNC_B128(lb, gb, 16);
            ASYNC_B128(lb, gb, 32); ASYNC_B128(lb, gb, 48);
            WAIT_ASYNC(8);    // my 8 loads of stage s complete (in-order), s+1 in flight
        } else {
            WAIT_ASYNC(0);
        }
        __syncthreads();      // stage s visible from all waves

        #pragma unroll
        for (int kk = 0; kk < 64; kk += 32) {
            // ---- preload ALL fragments of this k-half, then an uninterrupted WMMA chain ----
            v16bf bfrag[2], afrag[4];
            #pragma unroll
            for (int j = 0; j < 2; ++j) {
                // B: lane = col n = l15, K = kk + hi*16 .. +15 (contiguous)
                const bf16* p = &Bs[cur][wn + j * 16 + l15][kk + hi * 16];
                bf16x8 b0 = *(const bf16x8*)p;
                bf16x8 b1 = *(const bf16x8*)(p + 8);
                bfrag[j] = __builtin_shufflevector(b0, b1,
                    0,1,2,3,4,5,6,7,8,9,10,11,12,13,14,15);
            }
            #pragma unroll
            for (int i = 0; i < 4; ++i) {
                // A: lane = row m = l15, K = kk + hi*16 .. +15 (contiguous)
                const bf16* p = &As[cur][wm + i * 16 + l15][kk + hi * 16];
                bf16x8 a0 = *(const bf16x8*)p;
                bf16x8 a1 = *(const bf16x8*)(p + 8);
                afrag[i] = __builtin_shufflevector(a0, a1,
                    0,1,2,3,4,5,6,7,8,9,10,11,12,13,14,15);
            }
            #pragma unroll
            for (int i = 0; i < 4; ++i)
                #pragma unroll
                for (int j = 0; j < 2; ++j)
                    acc[i][j] = __builtin_amdgcn_wmma_f32_16x16x32_bf16(
                        false, afrag[i], false, bfrag[j], (short)0, acc[i][j], false, false);
        }
        __syncthreads();      // all waves done with buf[cur] before it is re-filled
    }

    // epilogue: D element e -> (m = e + hi*8, n = l15) within each 16x16 tile
    #pragma unroll
    for (int i = 0; i < 4; ++i) {
        #pragma unroll
        for (int j = 0; j < 2; ++j) {
            #pragma unroll
            for (int e = 0; e < 8; ++e) {
                const int row = m0 + wm + i * 16 + hi * 8 + e;
                const int col = n0 + wn + j * 16 + l15;
                const float v = acc[i][j][e];
                if (EPI == 0) {
                    ((bf16*)Cout)[(size_t)row * N + col] = (bf16)geluf(v);
                } else {
                    const size_t o = (size_t)row * N + col;
                    ((float*)Cout)[o] = v + resid[o];
                }
            }
        }
    }
}

// ---------------- launcher ----------------
extern "C" void kernel_launch(void* const* d_in, const int* in_sizes, int n_in,
                              void* d_out, int out_size, void* d_ws, size_t ws_size,
                              hipStream_t stream) {
    const float* x       = (const float*)d_in[0];
    const int*   lengths = (const int*)  d_in[1];
    const float* ln1w    = (const float*)d_in[2];
    const float* ln2w    = (const float*)d_in[3];
    const float* Wfc     = (const float*)d_in[4];
    const float* Wproj   = (const float*)d_in[5];
    float* out = (float*)d_out;

    char* ws = (char*)d_ws;
    size_t off = 0;
    auto alloc = [&](size_t bytes) { char* p = ws + off; off = (off + bytes + 255) & ~(size_t)255; return p; };

    float* x1       = (float*)alloc((size_t)MM * DD * sizeof(float));      //  64 MB
    bf16*  h2       = (bf16*) alloc((size_t)MM * DD * sizeof(bf16));       //  32 MB
    bf16*  WfcT     = (bf16*) alloc((size_t)HH * DD * sizeof(bf16));       //   8 MB
    bf16*  WprojT   = (bf16*) alloc((size_t)DD * HH * sizeof(bf16));       //   8 MB
    bf16*  act      = (bf16*) alloc((size_t)MM * HH * sizeof(bf16));       // 128 MB
    float* chunksum = (float*)alloc((size_t)BB * NC * DD * sizeof(float)); // 512 KB
    float* chunkpref= (float*)alloc((size_t)BB * NC * DD * sizeof(float)); // 512 KB

    transpose_cast_kernel<<<dim3(HH / 32, DD / 32), 256, 0, stream>>>(Wfc,   WfcT,   DD, HH);
    transpose_cast_kernel<<<dim3(DD / 32, HH / 32), 256, 0, stream>>>(Wproj, WprojT, HH, DD);

    ln1_scan_kernel<<<dim3(BB, NC), 256, 0, stream>>>(x, lengths, ln1w, x1, chunksum);
    chunk_prefix_kernel<<<(BB * DD + 255) / 256, 256, 0, stream>>>(chunksum, chunkpref);
    ln2_kernel<<<MM, 256, 0, stream>>>(x1, chunkpref, lengths, ln2w, h2);

    gemm_bf16_kernel<0><<<dim3(MM / 128, HH / 128), 256, 0, stream>>>(h2,  WfcT,   (void*)act, nullptr, MM, HH, DD);
    gemm_bf16_kernel<1><<<dim3(MM / 128, DD / 128), 256, 0, stream>>>(act, WprojT, (void*)out, x1,      MM, DD, HH);
}
